// Critic_1752346657343
// MI455X (gfx1250) — matrix operand order repaired and checked by reading, not compiled, not run on hardware
//
#include <hip/hip_runtime.h>

// CDNA5 / gfx1250: wave32, WMMA f16->f32 16x16x32 path.

typedef _Float16 half_t;
typedef __attribute__((ext_vector_type(2)))  _Float16 v2h;
typedef __attribute__((ext_vector_type(4)))  _Float16 v4h;
typedef __attribute__((ext_vector_type(8)))  _Float16 v8h;
typedef __attribute__((ext_vector_type(16))) _Float16 v16h;
typedef __attribute__((ext_vector_type(8)))  float    v8f;

#define HID    32
#define NODEF  4
#define NFACTC 3
#define WPB    8     // waves per 256-thread block
#define ETPW   8     // edge tiles per wave (weight-register reuse)
#define HTPW   8     // head tiles per wave

static __device__ __forceinline__ v8f wmma16(v16h a, v16h b, v8f c) {
    // D = A(16x32 f16) x B(32x16 f16) + C(16x16 f32)
    return __builtin_amdgcn_wmma_f32_16x16x32_f16(
        false, a, false, b, (short)0, c, false, false);
}

static __device__ __forceinline__ v16h cat8(v8h lo, v8h hi) {
    return __builtin_shufflevector(lo, hi, 0, 1, 2, 3, 4, 5, 6, 7,
                                   8, 9, 10, 11, 12, 13, 14, 15);
}

static __device__ __forceinline__ void st4(half_t* p, float a, float b,
                                           float c, float d) {
    v4h v;
    v.x = (half_t)a; v.y = (half_t)b; v.z = (half_t)c; v.w = (half_t)d;
    *reinterpret_cast<v4h*>(p) = v;   // ds_store_b64
}

// ---------------------------------------------------------------------------
__global__ void zero_kernel(float* __restrict__ xpp, float* __restrict__ out,
                            int nxpp, int nout) {
    int i = blockIdx.x * blockDim.x + threadIdx.x;
    if (i < nxpp) xpp[i] = 0.0f;
    if (i < nout) out[i] = 0.0f;
}

// ---------------------------------------------------------------------------
// Stage 1: EdgeConv.  One wave = ETPW x 16 edges; W1/W2 held in registers.
//   GEMM2's K dimension is stored K-permuted in LDS so each lane's two
//   N-tile outputs pack into one ds_store_b32; bw2 uses the inverse perm.
// ---------------------------------------------------------------------------
__global__ __launch_bounds__(256) void edgeconv_kernel(
    const float* __restrict__ x, const int* __restrict__ ei,
    const float* __restrict__ ea,
    const float* __restrict__ W1, const float* __restrict__ b1,
    const float* __restrict__ W2, const float* __restrict__ b2,
    float* __restrict__ xpp, int E, int ntiles)
{
    __shared__ alignas(16) half_t hbuf[WPB][16][HID];

    const int lane = threadIdx.x & 31;
    const int w    = threadIdx.x >> 5;
    const int hl   = lane >> 4;
    const int mi   = lane & 15;

    // ---- hoisted B operands (clamp-index + value-select: no exec branches)
    v16h bw1_0, bw1_1, bw2_0, bw2_1;
    #pragma unroll
    for (int e = 0; e < 16; ++e) {
        const int  k    = (e < 8) ? hl * 8 + e : 16 + hl * 8 + (e - 8);
        const int  kc   = (k < 2 * NODEF + 2) ? k : 0;
        const bool kval = (k < 2 * NODEF + 2);
        const float wa = W1[kc * HID + mi];
        const float wb = W1[kc * HID + 16 + mi];
        bw1_0[e] = kval ? (half_t)wa : (half_t)0;
        bw1_1[e] = kval ? (half_t)wb : (half_t)0;
        const int kp = (k >> 1) + 16 * (k & 1);    // inverse K-permutation
        bw2_0[e] = (half_t)W2[kp * HID + mi];
        bw2_1[e] = (half_t)W2[kp * HID + 16 + mi];
    }
    const float b1lo = b1[mi], b1hi = b1[16 + mi];
    const float b2lo = b2[mi], b2hi = b2[16 + mi];

    const float4* x4  = (const float4*)x;
    const float2* ea2 = (const float2*)ea;
    const int wavei = blockIdx.x * WPB + w;

    for (int it = 0; it < ETPW; ++it) {
        int tile = wavei * ETPW + it;
        const bool wact = (tile < ntiles);
        if (!wact) tile = 0;
        const int e0 = tile * 16;
        const int e  = e0 + mi;

        const int s_mine = ei[e];   // src id of row mi (valid in lanes 0..15)

        // ---- A1: per-edge features (half 0: K0..7 = x_src|x_dst, half 1: K8,9 = ea)
        v16h a1 = {};
        if (hl == 0) {
            const int d = ei[E + e];
            const float4 xs = x4[s_mine];
            const float4 xd = x4[d];
            a1[0] = (half_t)xs.x; a1[1] = (half_t)xs.y;
            a1[2] = (half_t)xs.z; a1[3] = (half_t)xs.w;
            a1[4] = (half_t)xd.x; a1[5] = (half_t)xd.y;
            a1[6] = (half_t)xd.z; a1[7] = (half_t)xd.w;
        } else {
            const float2 t = ea2[e];
            a1[0] = (half_t)t.x; a1[1] = (half_t)t.y;
        }

        v8f c0 = {}, c1 = {};
        c0 = wmma16(a1, bw1_0, c0);
        c1 = wmma16(a1, bw1_1, c1);

        // ---- relu(c + b1) -> LDS, K-permuted: cols (mi, 16+mi) adjacent
        #pragma unroll
        for (int r = 0; r < 8; ++r) {
            const int m = r + 8 * hl;
            v2h pk;
            pk.x = (half_t)fmaxf(c0[r] + b1lo, 0.0f);
            pk.y = (half_t)fmaxf(c1[r] + b1hi, 0.0f);
            *reinterpret_cast<v2h*>(&hbuf[w][m][2 * mi]) = pk;  // ds_store_b32
        }
        __builtin_amdgcn_wave_barrier();   // wave-private tile: DS in-order

        // ---- A2: two contiguous 8-half runs -> two ds_load_b128
        const v8h r0 = *(const v8h*)&hbuf[w][mi][8 * hl];
        const v8h r1 = *(const v8h*)&hbuf[w][mi][16 + 8 * hl];
        const v16h a2 = cat8(r0, r1);

        v8f d0 = {}, d1 = {};
        d0 = wmma16(a2, bw2_0, d0);
        d1 = wmma16(a2, bw2_1, d1);

        if (wact) {
            #pragma unroll
            for (int r = 0; r < 8; ++r) {
                const int m = r + 8 * hl;
                const int s = __shfl(s_mine, m, 32);   // src id of row m
                atomicAdd(&xpp[s * HID + mi],      d0[r] + b2lo);
                atomicAdd(&xpp[s * HID + 16 + mi], d1[r] + b2hi);
            }
        }
        __builtin_amdgcn_wave_barrier();
    }
}

// ---------------------------------------------------------------------------
// Stage 2: head.  One wave = HTPW x 16 rows; Wl held in registers.
// ---------------------------------------------------------------------------
__global__ __launch_bounds__(256) void critic_head_kernel(
    const float* __restrict__ x,  const float* __restrict__ xpp,
    const float* __restrict__ act,
    const int* __restrict__ esrc, const int* __restrict__ edst,
    const float* __restrict__ Wl, const float* __restrict__ bl,
    const float* __restrict__ Wv, const float* __restrict__ bv,
    float* __restrict__ out, int Bn, int NN, int NE, int gtiles)
{
    __shared__ alignas(16) half_t feat[WPB][16][96];

    const int lane = threadIdx.x & 31;
    const int w    = threadIdx.x >> 5;
    const int hl   = lane >> 4;
    const int mi   = lane & 15;
    const int ROWS = NE + NFACTC;

    // ---- hoisted B operands: Wl (73x32, K-padded to 96)
    v16h bw[3][2];
    #pragma unroll
    for (int c = 0; c < 3; ++c) {
        #pragma unroll
        for (int e = 0; e < 16; ++e) {
            const int  k    = c * 32 + ((e < 8) ? hl * 8 + e : 16 + hl * 8 + (e - 8));
            const int  kc   = (k < 73) ? k : 0;
            const bool kval = (k < 73);
            const float wa = Wl[kc * HID + mi];
            const float wb = Wl[kc * HID + 16 + mi];
            bw[c][0][e] = kval ? (half_t)wa : (half_t)0;
            bw[c][1][e] = kval ? (half_t)wb : (half_t)0;
        }
    }
    const float blo = bl[mi], bhi = bl[16 + mi];
    const float wv0 = Wv[mi], wv1 = Wv[16 + mi];
    const float bvv = bv[0];

    const float4* x4 = (const float4*)x;
    const int wavei = blockIdx.x * WPB + w;

    for (int it = 0; it < HTPW; ++it) {
        int gt = wavei * HTPW + it;
        const bool wact = (gt < Bn * gtiles);
        if (!wact) gt = 0;
        const int g    = gt / gtiles;
        const int tile = gt % gtiles;

        // ---- vectorized LDS fill: lane -> row mi, half -> 48-feature range
        const int  rr   = tile * 16 + mi;
        const bool vrow = (rr < ROWS);
        half_t* rowp = &feat[w][mi][0];
        if (vrow) {
            int s, d;
            if (rr < NE) { s = esrc[rr]; d = edst[rr]; }
            else         { s = d = NN - NFACTC + (rr - NE); }
            const int ns = g * NN + s, nd = g * NN + d;
            const float4* ps = (const float4*)&xpp[ns * HID];
            const float4* pd = (const float4*)&xpp[nd * HID];
            if (hl == 0) {
                const float4 xs = x4[ns];
                st4(rowp + 0, xs.x, xs.y, xs.z, xs.w);
                #pragma unroll
                for (int q = 0; q < 8; ++q) {
                    const float4 t = ps[q];
                    st4(rowp + 4 + 4 * q, t.x, t.y, t.z, t.w);
                }
                const float4 xd = x4[nd];
                st4(rowp + 36, xd.x, xd.y, xd.z, xd.w);
                const float4 t0 = pd[0], t1 = pd[1];
                st4(rowp + 40, t0.x, t0.y, t0.z, t0.w);
                st4(rowp + 44, t1.x, t1.y, t1.z, t1.w);
            } else {
                #pragma unroll
                for (int q = 2; q < 8; ++q) {
                    const float4 t = pd[q];
                    st4(rowp + 48 + 4 * (q - 2), t.x, t.y, t.z, t.w);
                }
                const float av = act[g * ROWS + rr];
                st4(rowp + 72, av, 0.0f, 0.0f, 0.0f);
                #pragma unroll
                for (int q = 0; q < 5; ++q)
                    st4(rowp + 76 + 4 * q, 0.0f, 0.0f, 0.0f, 0.0f);
            }
        } else {
            #pragma unroll
            for (int q = 0; q < 12; ++q)
                st4(rowp + 48 * hl + 4 * q, 0.0f, 0.0f, 0.0f, 0.0f);
        }
        __builtin_amdgcn_wave_barrier();   // wave-private tile: DS in-order

        // ---- 3 K-chunks x 2 N-tiles of WMMA
        v8f c0 = {}, c1 = {};
        #pragma unroll
        for (int c = 0; c < 3; ++c) {
            const v8h r0 = *(const v8h*)&feat[w][mi][c * 32 + 8 * hl];
            const v8h r1 = *(const v8h*)&feat[w][mi][c * 32 + 16 + 8 * hl];
            const v16h a = cat8(r0, r1);
            c0 = wmma16(a, bw[c][0], c0);
            c1 = wmma16(a, bw[c][1], c1);
        }

        // ---- fused 32->1 dot; single 32-lane reduction + one atomic
        float psum = 0.0f;
        #pragma unroll
        for (int r = 0; r < 8; ++r) {
            const int rr2 = tile * 16 + r + 8 * hl;
            const float p = fmaxf(c0[r] + blo, 0.0f) * wv0
                          + fmaxf(c1[r] + bhi, 0.0f) * wv1;
            psum += (rr2 < ROWS) ? p : 0.0f;
        }
        psum += __shfl_xor(psum, 1, 32);
        psum += __shfl_xor(psum, 2, 32);
        psum += __shfl_xor(psum, 4, 32);
        psum += __shfl_xor(psum, 8, 32);
        psum += __shfl_xor(psum, 16, 32);
        if (wact && lane == 0) {
            int nvalid = ROWS - tile * 16;
            nvalid = nvalid > 16 ? 16 : (nvalid < 0 ? 0 : nvalid);
            atomicAdd(&out[g], psum + (float)nvalid * bvv);
        }
        __builtin_amdgcn_wave_barrier();
    }
}

// ---------------------------------------------------------------------------
extern "C" void kernel_launch(void* const* d_in, const int* in_sizes, int n_in,
                              void* d_out, int out_size, void* d_ws, size_t ws_size,
                              hipStream_t stream) {
    const float* x    = (const float*)d_in[0];
    const int*   ei   = (const int*)  d_in[1];
    const float* ea   = (const float*)d_in[2];
    const float* act  = (const float*)d_in[3];
    const int*   esrc = (const int*)  d_in[4];
    const int*   edst = (const int*)  d_in[5];
    const float* W1   = (const float*)d_in[6];
    const float* b1   = (const float*)d_in[7];
    const float* W2   = (const float*)d_in[8];
    const float* b2   = (const float*)d_in[9];
    const float* Wl   = (const float*)d_in[10];
    const float* bl   = (const float*)d_in[11];
    const float* Wv   = (const float*)d_in[12];
    const float* bv   = (const float*)d_in[13];
    float* out = (float*)d_out;
    float* xpp = (float*)d_ws;            // [N][HID] accumulator

    const int N  = in_sizes[0] / NODEF;   // 65536 nodes total
    const int E  = in_sizes[2] / 2;       // 1,048,576 edges total
    const int NE = in_sizes[4];           // 1024 edges per graph
    const int Bn = out_size;              // 1024 graphs
    const int NN = N / Bn;                // 64 nodes per graph
    const int ROWS   = NE + NFACTC;       // 1027
    const int gtiles = (ROWS + 15) / 16;  // 65

    const int nxpp = N * HID;
    zero_kernel<<<(nxpp + 255) / 256, 256, 0, stream>>>(xpp, out, nxpp, out_size);

    const int etiles = (E + 15) / 16;                       // 65536 tiles
    const int ewaves = (etiles + ETPW - 1) / ETPW;          // 8192 waves
    edgeconv_kernel<<<(ewaves + WPB - 1) / WPB, 256, 0, stream>>>(
        x, ei, ea, W1, b1, W2, b2, xpp, E, etiles);

    const int htiles = Bn * gtiles;                         // 66560 tiles
    const int hwaves = (htiles + HTPW - 1) / HTPW;          // 8320 waves
    critic_head_kernel<<<(hwaves + WPB - 1) / WPB, 256, 0, stream>>>(
        x, xpp, act, esrc, edst, Wl, bl, Wv, bv, out, Bn, NN, NE, gtiles);
}
